// DSTGNN_77584289234973
// MI455X (gfx1250) — matrix operand order repaired and checked by reading, not compiled, not run on hardware
//
#include <hip/hip_runtime.h>
#include <hip/hip_bf16.h>
#include <math.h>

// ---------------------------------------------------------------------------
// DSTGNN forward for MI455X (gfx1250, wave32, WMMA).
// Big GEMMs -> v_wmma_f32_16x16x32_bf16 (f32 accumulate), bf16 staged via LDS.
// 128-thread blocks, 4 waves, wave tile 16x32 (A fragment reused across two
// B fragments -> 2 WMMA per stage, 3 ds_load_b128 per WMMA).
// Attention matrix (32x614x614) never materialized (on-the-fly A staging).
// Small-K contractions (K=10 weight-gen, K=3 Wh) stay on VALU.
// ---------------------------------------------------------------------------

#define NN   307
#define EE   10
#define DIM  3
#define HH   64
#define TT   12
#define GHD  256
#define BB   32
#define N2   614   // 2*NN

typedef __attribute__((ext_vector_type(16))) __bf16 v16bf;
typedef __attribute__((ext_vector_type(8)))  __bf16 v8bf;
typedef __attribute__((ext_vector_type(8)))  float  v8f;

// ---------------- WMMA fragment helpers (CDNA5 16x16x32 bf16 layout) -------
// A (16x32): lane = row (lane&15); half=lane>>4; K runs [half*8, half*8+7]
// and [16+half*8, 16+half*8+7].
__device__ __forceinline__ v16bf load_a_frag(const __bf16* rowp, int half) {
    const v8bf lo = *reinterpret_cast<const v8bf*>(rowp + half * 8);
    const v8bf hi = *reinterpret_cast<const v8bf*>(rowp + 16 + half * 8);
    v16bf r;
#pragma unroll
    for (int e = 0; e < 8; ++e) { r[e] = lo[e]; r[e + 8] = hi[e]; }
    return r;
}
// B (32x16) staged transposed in LDS as Bs[col][k]: lane = col (lane&15);
// K runs [half*8, half*8+7] and [16+half*8, 16+half*8+7] (same as A).
__device__ __forceinline__ v16bf load_b_frag(const __bf16* colp, int half) {
    const v8bf lo = *reinterpret_cast<const v8bf*>(colp + half * 8);
    const v8bf hi = *reinterpret_cast<const v8bf*>(colp + 16 + half * 8);
    v16bf r;
#pragma unroll
    for (int e = 0; e < 8; ++e) { r[e] = lo[e]; r[e + 8] = hi[e]; }
    return r;
}

__device__ __forceinline__ v8f wmma_bf16(v16bf a, v16bf b, v8f c) {
    return __builtin_amdgcn_wmma_f32_16x16x32_bf16(false, a, false, b,
                                                   (short)0, c, false, false);
}

__device__ __forceinline__ float apply_act(float v, int act) {
    if (act == 1) return (v > 0.f) ? v : (expf(v) - 1.f);             // elu
    if (act == 2) return 0.5f * v * (1.f + erff(v * 0.70710678118f)); // gelu
    if (act == 3) return fabsf(v);                                    // abs
    return v;
}

// ---------------- generic batched GEMM: C = A(MxK) * B(KxN) (+bias, act) ---
// block = 128 thr = 4 waves; block tile 32(M) x 64(N); wave tile 16x32.
__global__ __launch_bounds__(128) void k_gemm_bf16(
    const float* __restrict__ A, long long sAz, int lda,
    const float* __restrict__ B, long long sBz, int ldb,
    float* __restrict__ C, long long sCz, int ldc,
    int M, int Nn, int K, const float* __restrict__ bias, int act)
{
    __shared__ __align__(16) __bf16 As[32][48];
    __shared__ __align__(16) __bf16 Bs[64][48];
    const int tid = threadIdx.x;
    const int z = blockIdx.z;
    const int m0 = blockIdx.x * 32;
    const int n0 = blockIdx.y * 64;
    const float* Ab = A + (long long)z * sAz;
    const float* Bb = B + (long long)z * sBz;
    float* Cb = C + (long long)z * sCz;

    const int wave = tid >> 5, lane = tid & 31;
    const int mt = wave >> 1, nt = wave & 1;
    const int half = lane >> 4, l16 = lane & 15;
    const int ar = tid >> 2, ak = (tid & 3) * 8;    // A staging: 32 rows x 8 k
    const int bn = tid & 63, bk0 = (tid >> 6) * 16; // B staging: 64 cols x 16 k

    v8f acc0 = {0.f, 0.f, 0.f, 0.f, 0.f, 0.f, 0.f, 0.f};
    v8f acc1 = acc0;

    for (int k0 = 0; k0 < K; k0 += 32) {
        const bool kFull = (k0 + 32 <= K);
        // ---- stage A (row-major) -> As[row][k], packed bf16x8 store
        {
            const int m = m0 + ar;
            v8bf p;
            if (kFull && m < M) {
                const float4 f0 = *reinterpret_cast<const float4*>(
                    Ab + (long long)m * lda + (k0 + ak));
                const float4 f1 = *reinterpret_cast<const float4*>(
                    Ab + (long long)m * lda + (k0 + ak + 4));
                p[0] = (__bf16)f0.x; p[1] = (__bf16)f0.y;
                p[2] = (__bf16)f0.z; p[3] = (__bf16)f0.w;
                p[4] = (__bf16)f1.x; p[5] = (__bf16)f1.y;
                p[6] = (__bf16)f1.z; p[7] = (__bf16)f1.w;
                if (k0 + 64 <= K)  // prefetch next K slab
                    __builtin_prefetch(Ab + (long long)m * lda + (k0 + 32 + ak), 0, 0);
            } else {
#pragma unroll
                for (int i = 0; i < 8; ++i) {
                    const int k = k0 + ak + i;
                    p[i] = (m < M && k < K) ? (__bf16)Ab[(long long)m * lda + k]
                                            : (__bf16)0.f;
                }
            }
            *reinterpret_cast<v8bf*>(&As[ar][ak]) = p;
        }
        // ---- stage B (row-major KxN) transposed -> Bs[col][k], 2x bf16x8
        {
            const int n = n0 + bn;
            v8bf p0, p1;
            if (kFull && n < Nn) {
                const float* bp = Bb + (long long)(k0 + bk0) * ldb + n;
#pragma unroll
                for (int j = 0; j < 8; ++j) p0[j] = (__bf16)bp[(long long)j * ldb];
#pragma unroll
                for (int j = 0; j < 8; ++j) p1[j] = (__bf16)bp[(long long)(j + 8) * ldb];
                if (k0 + 64 <= K)
                    __builtin_prefetch(Bb + (long long)(k0 + 32 + bk0) * ldb + n, 0, 0);
            } else {
#pragma unroll
                for (int j = 0; j < 8; ++j) {
                    const int k = k0 + bk0 + j;
                    p0[j] = (n < Nn && k < K) ? (__bf16)Bb[(long long)k * ldb + n]
                                              : (__bf16)0.f;
                }
#pragma unroll
                for (int j = 0; j < 8; ++j) {
                    const int k = k0 + bk0 + 8 + j;
                    p1[j] = (n < Nn && k < K) ? (__bf16)Bb[(long long)k * ldb + n]
                                              : (__bf16)0.f;
                }
            }
            *reinterpret_cast<v8bf*>(&Bs[bn][bk0]) = p0;
            *reinterpret_cast<v8bf*>(&Bs[bn][bk0 + 8]) = p1;
        }
        __syncthreads();
        const v16bf af  = load_a_frag(&As[mt * 16 + l16][0], half);
        const v16bf bf0 = load_b_frag(&Bs[nt * 32 + l16][0], half);
        const v16bf bf1 = load_b_frag(&Bs[nt * 32 + 16 + l16][0], half);
        acc0 = wmma_bf16(af, bf0, acc0);
        acc1 = wmma_bf16(af, bf1, acc1);
        __syncthreads();
    }

#pragma unroll
    for (int s = 0; s < 2; ++s) {
        const v8f acc = s ? acc1 : acc0;
        const int n = n0 + nt * 32 + s * 16 + l16;
        if (n < Nn) {
            const float bv = bias ? bias[n] : 0.f;
#pragma unroll
            for (int g = 0; g < 8; ++g) {
                int m = m0 + mt * 16 + g + 8 * half;
                if (m < M)
                    Cb[(long long)m * ldc + n] = apply_act(acc[g] + bv, act);
            }
        }
    }
}

// ---------------- per-node GCN GEMM: out[b,node,:] = [xs|agg] @ W_node + b --
// grid.x = node; M=32 (batch), N=64, K=2c.
__global__ __launch_bounds__(128) void k_gcn_gemm(
    const float* __restrict__ xs, const float* __restrict__ agg, int c,
    const float* __restrict__ wt, const float* __restrict__ bt,
    float* __restrict__ out)
{
    __shared__ __align__(16) __bf16 As[32][48];
    __shared__ __align__(16) __bf16 Bs[64][48];
    const int node = blockIdx.x;
    const int tid = threadIdx.x;
    const int cc = 2 * c;
    const int wave = tid >> 5, lane = tid & 31;
    const int mt = wave >> 1, nt = wave & 1;
    const int half = lane >> 4, l16 = lane & 15;
    const int ar = tid >> 2, ak = (tid & 3) * 8;
    const int bn = tid & 63, bk0 = (tid >> 6) * 16;
    const float* wnode = wt + (long long)node * cc * HH;

    v8f acc0 = {0.f, 0.f, 0.f, 0.f, 0.f, 0.f, 0.f, 0.f};
    v8f acc1 = acc0;

    for (int k0 = 0; k0 < cc; k0 += 32) {
        // A gathered from [xs | agg] per node; packed store
        {
            const long long base = (long long)ar * NN + node;
            v8bf p;
#pragma unroll
            for (int i = 0; i < 8; ++i) {
                const int k = k0 + ak + i;
                float v = 0.f;
                if (k < cc)
                    v = (k < c) ? xs[base * c + k] : agg[base * c + (k - c)];
                p[i] = (__bf16)v;
            }
            *reinterpret_cast<v8bf*>(&As[ar][ak]) = p;
        }
        // B = W_node (cc x 64) transposed into Bs[col][k]
        {
            v8bf p0, p1;
            if (k0 + 32 <= cc) {
                const float* bp = wnode + (long long)(k0 + bk0) * HH + bn;
#pragma unroll
                for (int j = 0; j < 8; ++j) p0[j] = (__bf16)bp[(long long)j * HH];
#pragma unroll
                for (int j = 0; j < 8; ++j) p1[j] = (__bf16)bp[(long long)(j + 8) * HH];
            } else {
#pragma unroll
                for (int j = 0; j < 8; ++j) {
                    const int k = k0 + bk0 + j;
                    p0[j] = (k < cc) ? (__bf16)wnode[(long long)k * HH + bn]
                                     : (__bf16)0.f;
                }
#pragma unroll
                for (int j = 0; j < 8; ++j) {
                    const int k = k0 + bk0 + 8 + j;
                    p1[j] = (k < cc) ? (__bf16)wnode[(long long)k * HH + bn]
                                     : (__bf16)0.f;
                }
            }
            *reinterpret_cast<v8bf*>(&Bs[bn][bk0]) = p0;
            *reinterpret_cast<v8bf*>(&Bs[bn][bk0 + 8]) = p1;
        }
        __syncthreads();
        const v16bf af  = load_a_frag(&As[mt * 16 + l16][0], half);
        const v16bf bf0 = load_b_frag(&Bs[nt * 32 + l16][0], half);
        const v16bf bf1 = load_b_frag(&Bs[nt * 32 + 16 + l16][0], half);
        acc0 = wmma_bf16(af, bf0, acc0);
        acc1 = wmma_bf16(af, bf1, acc1);
        __syncthreads();
    }

#pragma unroll
    for (int s = 0; s < 2; ++s) {
        const v8f acc = s ? acc1 : acc0;
        const int o = nt * 32 + s * 16 + l16;
        const float bv = bt[node * HH + o];
#pragma unroll
        for (int g = 0; g < 8; ++g) {
            int m = mt * 16 + g + 8 * half;  // batch row
            out[((long long)m * NN + node) * HH + o] = acc[g] + bv;
        }
    }
}

// ---------------- fused GAT:  hcat = elu(concat-split( att @ Wh )) ---------
// att built on the fly from column-softmax stats; never materialized.
__global__ __launch_bounds__(128) void k_gat_hp(
    const float* __restrict__ wh, const float* __restrict__ s1,
    const float* __restrict__ s2, const float* __restrict__ cm,
    const float* __restrict__ cis, const unsigned char* __restrict__ adj,
    float* __restrict__ hcat)
{
    __shared__ __align__(16) __bf16 As[32][48];
    __shared__ __align__(16) __bf16 Bs[64][48];
    const int tid = threadIdx.x;
    const int b = blockIdx.z;
    const int i0 = blockIdx.x * 32;
    const int o0 = blockIdx.y * 64;
    const int wave = tid >> 5, lane = tid & 31;
    const int mt = wave >> 1, nt = wave & 1;
    const int half = lane >> 4, l16 = lane & 15;
    const int ar = tid >> 2, ak = (tid & 3) * 8;
    const int bn = tid & 63, bk0 = (tid >> 6) * 16;

    const float* whb = wh + (long long)b * N2 * GHD;
    const float* s1b = s1 + (long long)b * N2;
    const float* s2b = s2 + (long long)b * N2;
    const float* cmb = cm + (long long)b * N2;
    const float* cib = cis + (long long)b * N2;

    v8f acc0 = {0.f, 0.f, 0.f, 0.f, 0.f, 0.f, 0.f, 0.f};
    v8f acc1 = acc0;
    const int r = i0 + ar;
    const float s1r = (r < N2) ? s1b[r] : 0.f;

    for (int k0 = 0; k0 < N2; k0 += 32) {
        // A tile = attention values computed on the fly; packed store
        {
            v8bf p;
#pragma unroll
            for (int i = 0; i < 8; ++i) {
                const int j = k0 + ak + i;
                float v = 0.f;
                if (r < N2 && j < N2 && adj[(long long)r * N2 + j]) {
                    float e = s1r + s2b[j];
                    e = (e > 0.f) ? e : 0.2f * e;            // leaky_relu 0.2
                    v = expf(e - cmb[j]) * cib[j];           // softmax axis=1
                }
                p[i] = (__bf16)v;
            }
            *reinterpret_cast<v8bf*>(&As[ar][ak]) = p;
        }
        // B = Wh[b] (614 x 256) transposed into Bs[col][k]
        {
            v8bf p0, p1;
            if (k0 + 32 <= N2) {
                const float* bp = whb + (long long)(k0 + bk0) * GHD + (o0 + bn);
#pragma unroll
                for (int j = 0; j < 8; ++j) p0[j] = (__bf16)bp[(long long)j * GHD];
#pragma unroll
                for (int j = 0; j < 8; ++j) p1[j] = (__bf16)bp[(long long)(j + 8) * GHD];
            } else {
#pragma unroll
                for (int j = 0; j < 8; ++j) {
                    const int k = k0 + bk0 + j;
                    p0[j] = (k < N2) ? (__bf16)whb[(long long)k * GHD + (o0 + bn)]
                                     : (__bf16)0.f;
                }
#pragma unroll
                for (int j = 0; j < 8; ++j) {
                    const int k = k0 + bk0 + 8 + j;
                    p1[j] = (k < N2) ? (__bf16)whb[(long long)k * GHD + (o0 + bn)]
                                     : (__bf16)0.f;
                }
            }
            *reinterpret_cast<v8bf*>(&Bs[bn][bk0]) = p0;
            *reinterpret_cast<v8bf*>(&Bs[bn][bk0 + 8]) = p1;
        }
        __syncthreads();
        const v16bf af  = load_a_frag(&As[mt * 16 + l16][0], half);
        const v16bf bf0 = load_b_frag(&Bs[nt * 32 + l16][0], half);
        const v16bf bf1 = load_b_frag(&Bs[nt * 32 + 16 + l16][0], half);
        acc0 = wmma_bf16(af, bf0, acc0);
        acc1 = wmma_bf16(af, bf1, acc1);
        __syncthreads();
    }

#pragma unroll
    for (int s = 0; s < 2; ++s) {
        const v8f acc = s ? acc1 : acc0;
        const int o = o0 + nt * 32 + s * 16 + l16;
#pragma unroll
        for (int g = 0; g < 8; ++g) {
            int i = i0 + mt * 16 + g + 8 * half;
            if (i < N2) {
                float v = acc[g];
                v = (v > 0.f) ? v : (expf(v) - 1.f);  // elu
                long long dst = (i < NN)
                    ? (((long long)b * NN + i) * 512 + o)
                    : (((long long)b * NN + (i - NN)) * 512 + GHD + o);
                hcat[dst] = v;
            }
        }
    }
}

// ---------------- small VALU kernels ---------------------------------------
__global__ void k_zero(float* __restrict__ p, long long n) {
    long long i = (long long)blockIdx.x * blockDim.x + threadIdx.x;
    if (i < n) p[i] = 0.f;
}

__global__ void k_ne_ln(const float* __restrict__ node_emb,
                        const float* __restrict__ te,
                        const float* __restrict__ g, const float* __restrict__ b,
                        float* __restrict__ ne) {
    int n = blockIdx.x * blockDim.x + threadIdx.x;
    if (n >= NN) return;
    float v[EE], m = 0.f;
#pragma unroll
    for (int d = 0; d < EE; ++d) { v[d] = node_emb[n * EE + d] + te[d]; m += v[d]; }
    m *= (1.f / EE);
    float var = 0.f;
#pragma unroll
    for (int d = 0; d < EE; ++d) { float t = v[d] - m; var += t * t; }
    var *= (1.f / EE);
    float rs = rsqrtf(var + 1e-12f);
#pragma unroll
    for (int d = 0; d < EE; ++d)
        ne[n * EE + d] = (v[d] - m) * rs * g[d] + b[d];
}

__global__ __launch_bounds__(512) void k_emb_softmax(
    const float* __restrict__ ne, float* __restrict__ emb) {
    __shared__ float sm[512];
    const int row = blockIdx.x, tid = threadIdx.x;
    float e = -3.0e38f;
    if (tid < NN) {
        float acc = 0.f;
#pragma unroll
        for (int d = 0; d < EE; ++d) acc += ne[row * EE + d] * ne[tid * EE + d];
        e = acc;
    }
    sm[tid] = e;
    __syncthreads();
    for (int s = 256; s > 0; s >>= 1) {
        if (tid < s) sm[tid] = fmaxf(sm[tid], sm[tid + s]);
        __syncthreads();
    }
    const float mx = sm[0];
    __syncthreads();
    const float ex = (tid < NN) ? expf(e - mx) : 0.f;
    sm[tid] = ex;
    __syncthreads();
    for (int s = 256; s > 0; s >>= 1) {
        if (tid < s) sm[tid] += sm[tid + s];
        __syncthreads();
    }
    if (tid < NN) emb[(long long)row * NN + tid] = ex / sm[0];
}

// WT[n][j] = sum_d ne[n,d] * w_pool[d*J + j]   (K=10: VALU beats padded WMMA)
__global__ void k_wgen(const float* __restrict__ ne,
                       const float* __restrict__ wp,
                       float* __restrict__ wt, int J) {
    int n = blockIdx.x;
    int j = blockIdx.y * blockDim.x + threadIdx.x;
    if (j >= J) return;
    float acc = 0.f;
#pragma unroll
    for (int d = 0; d < EE; ++d) acc += ne[n * EE + d] * wp[(long long)d * J + j];
    wt[(long long)n * J + j] = acc;
}

__global__ void k_bgen(const float* __restrict__ ne,
                       const float* __restrict__ bp, float* __restrict__ bt) {
    int i = blockIdx.x * blockDim.x + threadIdx.x;
    if (i >= NN * HH) return;
    int n = i / HH, o = i % HH;
    float acc = 0.f;
#pragma unroll
    for (int d = 0; d < EE; ++d) acc += ne[n * EE + d] * bp[d * HH + o];
    bt[i] = acc;
}

// xs[b,n,k] = k<c1 ? x[b,n,k] : (z ? z*state : state)
__global__ void k_concat(const float* __restrict__ x, long long xbs, int c1,
                         const float* __restrict__ st,
                         const float* __restrict__ z,
                         float* __restrict__ xs, int c) {
    long long idx = (long long)blockIdx.x * blockDim.x + threadIdx.x;
    long long total = (long long)BB * NN * c;
    if (idx >= total) return;
    int k = (int)(idx % c);
    long long bn = idx / c;
    int n = (int)(bn % NN);
    int b = (int)(bn / NN);
    float v;
    if (k < c1) v = x[(long long)b * xbs + (long long)n * c1 + k];
    else {
        long long j = ((long long)b * NN + n) * HH + (k - c1);
        v = z ? z[j] * st[j] : st[j];
    }
    xs[idx] = v;
}

__global__ void k_sigmoid(float* __restrict__ p, long long n) {
    long long i = (long long)blockIdx.x * blockDim.x + threadIdx.x;
    if (i < n) p[i] = 1.f / (1.f + expf(-p[i]));
}

// state = r*state + (1-r)*tanh(hc)
__global__ void k_gru_out(const float* __restrict__ r,
                          const float* __restrict__ hc,
                          float* __restrict__ state, long long n) {
    long long i = (long long)blockIdx.x * blockDim.x + threadIdx.x;
    if (i < n) {
        float rv = r[i];
        state[i] = rv * state[i] + (1.f - rv) * tanhf(hc[i]);
    }
}

// Wh[b,i,o] = sum_{d<3} h[b,i,d] * W[d,o];  i<NN -> prev slice, else cur.
__global__ void k_wh(const float* __restrict__ cur,
                     const float* __restrict__ prev, long long bs,
                     const float* __restrict__ W, float* __restrict__ wh) {
    long long idx = (long long)blockIdx.x * blockDim.x + threadIdx.x;
    long long total = (long long)BB * N2 * GHD;
    if (idx >= total) return;
    int o = (int)(idx & (GHD - 1));
    long long bi = idx >> 8;
    int i = (int)(bi % N2);
    int b = (int)(bi / N2);
    const float* src = (i < NN) ? prev : cur;
    const float* h = src + (long long)b * bs + (long long)(i % NN) * DIM;
    float acc = 0.f;
#pragma unroll
    for (int d = 0; d < DIM; ++d) acc += h[d] * W[d * GHD + o];
    wh[idx] = acc;
}

__global__ void k_scores(const float* __restrict__ wh,
                         const float* __restrict__ a,
                         float* __restrict__ s1, float* __restrict__ s2) {
    int idx = blockIdx.x * blockDim.x + threadIdx.x;
    if (idx >= BB * N2) return;
    const float* row = wh + (long long)idx * GHD;
    float a1 = 0.f, a2 = 0.f;
    for (int o = 0; o < GHD; ++o) { a1 += row[o] * a[o]; a2 += row[o] * a[GHD + o]; }
    s1[idx] = a1; s2[idx] = a2;
}

// per (b,j): column max / 1/sum for softmax over i (axis=1)
__global__ void k_colstat(const float* __restrict__ s1,
                          const float* __restrict__ s2,
                          const unsigned char* __restrict__ adj,
                          float* __restrict__ cm, float* __restrict__ cis) {
    int idx = blockIdx.x * blockDim.x + threadIdx.x;
    if (idx >= BB * N2) return;
    int j = idx % N2, b = idx / N2;
    const float* s1b = s1 + (long long)b * N2;
    const float sj = s2[(long long)b * N2 + j];
    float mx = -9.0e15f;
    for (int i = 0; i < N2; ++i) {
        if (adj[(long long)i * N2 + j]) {
            float e = s1b[i] + sj;
            e = (e > 0.f) ? e : 0.2f * e;
            mx = fmaxf(mx, e);
        }
    }
    float sum = 0.f;
    for (int i = 0; i < N2; ++i) {
        float e = adj[(long long)i * N2 + j]
                    ? ((s1b[i] + sj > 0.f) ? s1b[i] + sj : 0.2f * (s1b[i] + sj))
                    : -9.0e15f;
        sum += expf(e - mx);
    }
    cm[idx] = mx;
    cis[idx] = 1.f / sum;
}

// inner = LN(state + att) over H, eps 1e-5
__global__ void k_addln(const float* __restrict__ state,
                        const float* __restrict__ att,
                        const float* __restrict__ g, const float* __restrict__ b,
                        float* __restrict__ out) {
    int idx = blockIdx.x * blockDim.x + threadIdx.x;
    if (idx >= BB * NN) return;
    const long long base = (long long)idx * HH;
    float m = 0.f;
    for (int h = 0; h < HH; ++h) m += state[base + h] + att[base + h];
    m *= (1.f / HH);
    float var = 0.f;
    for (int h = 0; h < HH; ++h) {
        float t = state[base + h] + att[base + h] - m;
        var += t * t;
    }
    var *= (1.f / HH);
    float rs = rsqrtf(var + 1e-5f);
    for (int h = 0; h < HH; ++h)
        out[base + h] = (state[base + h] + att[base + h] - m) * rs * g[h] + b[h];
}

// out[b,o,n] = conv_b[o] + sum_h LN(s1)[h]*cw[o,0,h] + LN(base)[h]*cw[o,1,h]
__global__ void k_final(const float* __restrict__ s1buf,
                        const float* __restrict__ base,
                        const float* __restrict__ g, const float* __restrict__ bt,
                        const float* __restrict__ cw, const float* __restrict__ cb,
                        float* __restrict__ out) {
    int idx = blockIdx.x * blockDim.x + threadIdx.x;
    if (idx >= BB * NN) return;
    int n = idx % NN, b = idx / NN;
    const float* r1 = s1buf + (long long)idx * HH;
    const float* r2 = base + (long long)idx * HH;
    float m1 = 0.f, m2 = 0.f;
    for (int h = 0; h < HH; ++h) { m1 += r1[h]; m2 += r2[h]; }
    m1 *= (1.f / HH); m2 *= (1.f / HH);
    float v1 = 0.f, v2 = 0.f;
    for (int h = 0; h < HH; ++h) {
        float t1 = r1[h] - m1, t2 = r2[h] - m2;
        v1 += t1 * t1; v2 += t2 * t2;
    }
    float rs1 = rsqrtf(v1 * (1.f / HH) + 1e-12f);
    float rs2 = rsqrtf(v2 * (1.f / HH) + 1e-12f);
    for (int o = 0; o < TT; ++o) {
        float acc = cb[o];
        for (int h = 0; h < HH; ++h) {
            float l1 = (r1[h] - m1) * rs1 * g[h] + bt[h];
            float l2 = (r2[h] - m2) * rs2 * g[h] + bt[h];
            acc += l1 * cw[(o * 2 + 0) * HH + h] + l2 * cw[(o * 2 + 1) * HH + h];
        }
        out[((long long)b * TT + o) * NN + n] = acc;
    }
}

// ---------------------------------------------------------------------------
// Host-side sequencing.
// Input order (jax pytree: top-level dict insertion order, nested dicts by
// sorted keys, lists in order):
//  0 source  1 adj  2 conv_b  3 conv_w
//  4..75  gats[12] x {W, a, fc1_b, fc1_w, fc2_b, fc2_w}
//  76..87 gru0 {r,u,z} x {b_pool, ln_b, ln_g, w_pool}
//  88..111 grus[2] x {r,u,z} x {b_pool, ln_b, ln_g, w_pool}
//  112 node_emb  113..136 norms[12] x {b,g}  137,138 out_ln {b,g}  139 time_emb
// ---------------------------------------------------------------------------
extern "C" void kernel_launch(void* const* d_in, const int* in_sizes, int n_in,
                              void* d_out, int out_size, void* d_ws, size_t ws_size,
                              hipStream_t stream) {
    (void)in_sizes; (void)n_in; (void)out_size; (void)ws_size;
    auto F = [&](int i) { return (const float*)d_in[i]; };
    const float* SRC  = F(0);
    const unsigned char* ADJ = (const unsigned char*)d_in[1];
    const float* CONVB = F(2);
    const float* CONVW = F(3);
    const float* NODE = F(112);
    const float* OLN_B = F(137);
    const float* OLN_G = F(138);
    const float* TE = F(139);
    float* OUT = (float*)d_out;

    // workspace carve-out
    float* wsf = (float*)d_ws;
    long long off = 0;
    auto alloc = [&](long long nf) { float* p = wsf + off; off += nf; return p; };
    float* NE    = alloc((long long)NN * EE);
    float* EMB   = alloc((long long)NN * NN);
    float* BT    = alloc((long long)NN * HH);
    float* WT    = alloc((long long)NN * 256 * HH);
    float* XS    = alloc((long long)BB * NN * 128);
    float* AGG   = alloc((long long)BB * NN * 128);
    float* GZ    = alloc((long long)BB * NN * HH);
    float* GR    = alloc((long long)BB * NN * HH);
    float* GHB   = alloc((long long)BB * NN * HH);
    float* STATE = alloc((long long)BB * NN * HH);
    float* S2A   = alloc((long long)BB * NN * HH);
    float* S2B   = alloc((long long)BB * NN * HH);
    float* INNER = alloc((long long)BB * NN * HH);
    float* ATTO  = alloc((long long)BB * NN * HH);
    float* WH    = alloc((long long)BB * N2 * GHD);
    float* SC1   = alloc((long long)BB * N2);
    float* SC2   = alloc((long long)BB * N2);
    float* CM    = alloc((long long)BB * N2);
    float* CIS   = alloc((long long)BB * N2);
    float* HCAT  = alloc((long long)BB * NN * 512);
    float* Y1    = alloc((long long)BB * NN * GHD);

    const long long NST = (long long)BB * NN * HH;
    auto zero = [&](float* p, long long n) {
        k_zero<<<dim3((unsigned)((n + 255) / 256)), 256, 0, stream>>>(p, n);
    };
    zero(STATE, NST); zero(S2A, NST); zero(S2B, NST);

    // one GCN call: ne/emb/W/b generation + graph aggregation + node GEMM
    auto run_gcn = [&](int pbase, int c, const float* xs, float* out,
                       const float* te_t) {
        const float* b_pool = F(pbase + 0);
        const float* lnb    = F(pbase + 1);
        const float* lng    = F(pbase + 2);
        const float* w_pool = F(pbase + 3);
        k_ne_ln<<<1, 512, 0, stream>>>(NODE, te_t, lng, lnb, NE);
        k_emb_softmax<<<NN, 512, 0, stream>>>(NE, EMB);
        const int J = 2 * c * HH;
        k_wgen<<<dim3(NN, (J + 255) / 256), 256, 0, stream>>>(NE, w_pool, WT, J);
        k_bgen<<<(NN * HH + 255) / 256, 256, 0, stream>>>(NE, b_pool, BT);
        // AGG[b] = EMB(307x307) @ xs[b](307xc)  -- WMMA
        k_gemm_bf16<<<dim3((NN + 31) / 32, (c + 63) / 64, BB), 128, 0, stream>>>(
            EMB, 0LL, NN, xs, (long long)NN * c, c, AGG, (long long)NN * c, c,
            NN, c, NN, nullptr, 0);
        k_gcn_gemm<<<NN, 128, 0, stream>>>(xs, AGG, c, WT, BT, out);
    };

    const long long ELTS = NST;
    const unsigned EB = (unsigned)((ELTS + 255) / 256);

    for (int ti = 0; ti < TT; ++ti) {
        const float* te_t = TE + ti * EE;
        const float* xsrc = SRC + (long long)ti * NN * DIM;
        const long long sbs = (long long)TT * NN * DIM;
        const int c0 = DIM + HH;   // 67
        const long long cat0 = (long long)BB * NN * c0;

        // ---- gru0 on x = source[:, ti] ----
        k_concat<<<(unsigned)((cat0 + 255) / 256), 256, 0, stream>>>(
            xsrc, sbs, DIM, STATE, nullptr, XS, c0);
        run_gcn(84, c0, XS, GZ, te_t);                 // z
        k_sigmoid<<<EB, 256, 0, stream>>>(GZ, ELTS);
        run_gcn(76, c0, XS, GR, te_t);                 // r
        k_sigmoid<<<EB, 256, 0, stream>>>(GR, ELTS);
        k_concat<<<(unsigned)((cat0 + 255) / 256), 256, 0, stream>>>(
            xsrc, sbs, DIM, STATE, GZ, XS, c0);
        run_gcn(80, c0, XS, GHB, te_t);                // u (candidate)
        k_gru_out<<<EB, 256, 0, stream>>>(GR, GHB, STATE, ELTS);

        // ---- GAT ----
        const int tp = (ti > 0) ? (ti - 1) : 0;
        const int gb = 4 + ti * 6;
        const float* gW   = F(gb + 0);
        const float* gA   = F(gb + 1);
        const float* f1b  = F(gb + 2);
        const float* f1w  = F(gb + 3);
        const float* f2b  = F(gb + 4);
        const float* f2w  = F(gb + 5);
        {
            const long long nwh = (long long)BB * N2 * GHD;
            k_wh<<<(unsigned)((nwh + 255) / 256), 256, 0, stream>>>(
                xsrc, SRC + (long long)tp * NN * DIM, sbs, gW, WH);
        }
        k_scores<<<(BB * N2 + 255) / 256, 256, 0, stream>>>(WH, gA, SC1, SC2);
        k_colstat<<<(BB * N2 + 255) / 256, 256, 0, stream>>>(SC1, SC2, ADJ, CM, CIS);
        k_gat_hp<<<dim3((N2 + 31) / 32, GHD / 64, BB), 128, 0, stream>>>(
            WH, SC1, SC2, CM, CIS, ADJ, HCAT);
        // fc1: (BB*NN, 512) @ (512, 256) + bias, GELU
        k_gemm_bf16<<<dim3((BB * NN + 31) / 32, GHD / 64, 1), 128, 0, stream>>>(
            HCAT, 0LL, 512, f1w, 0LL, GHD, Y1, 0LL, GHD,
            BB * NN, GHD, 512, f1b, 2);
        // fc2: (BB*NN, 256) @ (256, 64) + bias, |.| (elu(|y|)=|y|)
        k_gemm_bf16<<<dim3((BB * NN + 31) / 32, 1, 1), 128, 0, stream>>>(
            Y1, 0LL, GHD, f2w, 0LL, HH, ATTO, 0LL, HH,
            BB * NN, HH, GHD, f2b, 3);
        k_addln<<<(BB * NN + 255) / 256, 256, 0, stream>>>(
            STATE, ATTO, F(113 + ti * 2 + 1), F(113 + ti * 2), INNER);

        // ---- inner GRU (grus[ti%2]) on inner[ti], fused into same loop ----
        float* Si = (ti & 1) ? S2B : S2A;
        const int pg = 88 + (ti & 1) * 12;
        const int c1 = 2 * HH;  // 128
        const long long cat1 = (long long)BB * NN * c1;
        k_concat<<<(unsigned)((cat1 + 255) / 256), 256, 0, stream>>>(
            INNER, (long long)NN * HH, HH, Si, nullptr, XS, c1);
        run_gcn(pg + 8, c1, XS, GZ, te_t);             // z
        k_sigmoid<<<EB, 256, 0, stream>>>(GZ, ELTS);
        run_gcn(pg + 0, c1, XS, GR, te_t);             // r
        k_sigmoid<<<EB, 256, 0, stream>>>(GR, ELTS);
        k_concat<<<(unsigned)((cat1 + 255) / 256), 256, 0, stream>>>(
            INNER, (long long)NN * HH, HH, Si, GZ, XS, c1);
        run_gcn(pg + 4, c1, XS, GHB, te_t);            // u
        k_gru_out<<<EB, 256, 0, stream>>>(GR, GHB, Si, ELTS);
    }

    // out channels [-2:] of stack([s0, s1, base]) -> (states[1], base_state)
    k_final<<<(BB * NN + 255) / 256, 256, 0, stream>>>(
        S2B, STATE, OLN_G, OLN_B, CONVW, CONVB, OUT);
}
